// LSTM_33964601376934
// MI455X (gfx1250) — matrix operand order repaired and checked by reading, not compiled, not run on hardware
//
#include <hip/hip_runtime.h>
#include <hip/hip_bf16.h>

typedef __attribute__((ext_vector_type(16))) _Float16 v16h;
typedef __attribute__((ext_vector_type(8)))  _Float16 v8h;
typedef __attribute__((ext_vector_type(8)))  float    v8f;

// ---------------- problem constants ----------------
#define B_TOT   8192
#define T_STEPS 512
#define IN_DIM  7

// fragment table: 60 B-matrix fragments, each 32 lanes x 16 halves = 1 KB
//  idx  0.. 3 : [Wih1|Whh1] combined (60 x 22: K0..6 = x, K7..21 = h1)  nt 0..3
//  idx  4.. 8 : Wih2 (80x15)  K-shifted by 7 (zero rows K<7)            nt 0..4
//  idx  9..13 : Whh2 (80x20)                                            nt 0..4
//  idx 14..21 : Wih3 (120x20)                                           nt 0..7
//  idx 22..29 : Whh3 (120x30)                                           nt 0..7
//  idx 30..39 : Wih4 (160x30)                                           nt 0..9
//  idx 40..59 : Whh4 (160x40) kb 0..1  (idx = 40 + kb*10 + nt)
constexpr int FB_L1  = 0;
constexpr int FB_IH2 = 4,  FB_HH2 = 9;
constexpr int FB_IH3 = 14, FB_HH3 = 22;
constexpr int FB_IH4 = 30, FB_HH4 = 40;
constexpr int NFRAG  = 60;

constexpr int WS_FRAG_BYTES = NFRAG * 512 * 2;          // 61440
constexpr int WS_BIAS_OFF   = WS_FRAG_BYTES;            // 432 f32 (64+80+128+160)
constexpr int WS_WL_OFF     = WS_BIAS_OFF + 432 * 4;    // 41 f32 (Wl[40], bl)
constexpr int WS_TOTAL      = 63344;                    // 16B multiple >= 63332

// gate scratch: column-major [160 cols][GSTR], stride padded to 20 dwords (80 B)
// -> b128-aligned per-column stores; lane->bank map (20L mod 64) conflict-free
constexpr int GSTR = 20;

constexpr int WAVES_PER_BLK = 4;
// per-wave LDS: f16 h-state 5120 B + f32 c-state 6720 B + gate scratch 12800 B
constexpr int PW_BYTES = 5120 + 6720 + 160 * GSTR * 4;  // 24640

// ---------------- prep kernel 1: pack weights into WMMA B-fragments -------
// B layout packed here (16-bit, 32x16, contiguous K): lane L = column (L&15); K = kb*32 + (L<16?0:16) + e
__global__ void lstm_pack_frags(const float* __restrict__ Wih1, const float* __restrict__ Whh1,
                                const float* __restrict__ Wih2, const float* __restrict__ Whh2,
                                const float* __restrict__ Wih3, const float* __restrict__ Whh3,
                                const float* __restrict__ Wih4, const float* __restrict__ Whh4,
                                _Float16* __restrict__ frag) {
  const int f    = blockIdx.x;        // 0..59
  const int lane = threadIdx.x;       // 0..31
  const int col  = lane & 15;
  const int khi  = (lane & 16) ? 16 : 0;
  #pragma unroll
  for (int e = 0; e < 16; ++e) {
    float v = 0.0f;
    if (f < 4) {                                   // combined [x|h1] layer-1 block
      const int n = f * 16 + col, k = khi + e;
      if (n < 60) {
        if (k < 7)       v = Wih1[n * 7 + k];
        else if (k < 22) v = Whh1[n * 15 + (k - 7)];
      }
    } else if (f < 9) {                            // Wih2, K-shifted by 7
      const int n = (f - 4) * 16 + col, k = khi + e;
      if (n < 80 && k >= 7 && k < 22) v = Wih2[n * 15 + (k - 7)];
    } else if (f < 14) {                           // Whh2
      const int n = (f - 9) * 16 + col, k = khi + e;
      if (n < 80 && k < 20) v = Whh2[n * 20 + k];
    } else if (f < 22) {                           // Wih3
      const int n = (f - 14) * 16 + col, k = khi + e;
      if (n < 120 && k < 20) v = Wih3[n * 20 + k];
    } else if (f < 30) {                           // Whh3
      const int n = (f - 22) * 16 + col, k = khi + e;
      if (n < 120 && k < 30) v = Whh3[n * 30 + k];
    } else if (f < 40) {                           // Wih4
      const int n = (f - 30) * 16 + col, k = khi + e;
      if (n < 160 && k < 30) v = Wih4[n * 30 + k];
    } else {                                       // Whh4, kb 0..1
      const int kb = (f - 40) / 10, nt = (f - 40) % 10;
      const int n = nt * 16 + col, k = kb * 32 + khi + e;
      if (n < 160 && k < 40) v = Whh4[n * 40 + k];
    }
    frag[f * 512 + lane * 16 + e] = (_Float16)v;
  }
}

// ---------------- prep kernel 2: combined biases + final linear -----------
__global__ void lstm_pack_bias(const float* __restrict__ bi1, const float* __restrict__ bh1,
                               const float* __restrict__ bi2, const float* __restrict__ bh2,
                               const float* __restrict__ bi3, const float* __restrict__ bh3,
                               const float* __restrict__ bi4, const float* __restrict__ bh4,
                               const float* __restrict__ Wl,  const float* __restrict__ bl,
                               float* __restrict__ biasOut, float* __restrict__ wlOut) {
  const int tid = threadIdx.x;
  if (tid < 432) {
    int off, n4h; const float* bi; const float* bh;
    if      (tid <  64) { off = 0;   n4h = 60;  bi = bi1; bh = bh1; }
    else if (tid < 144) { off = 64;  n4h = 80;  bi = bi2; bh = bh2; }
    else if (tid < 272) { off = 144; n4h = 120; bi = bi3; bh = bh3; }
    else                { off = 272; n4h = 160; bi = bi4; bh = bh4; }
    const int j = tid - off;
    biasOut[tid] = (j < n4h) ? (bi[j] + bh[j]) : 0.0f;
  } else if (tid < 473) {
    const int k = tid - 432;
    wlOut[k] = (k < 40) ? Wl[k] : bl[0];
  }
}

// ---------------- device helpers -----------------------------------------
// Build 16-bit A fragment (16x32) from f16 LDS row `rowp` (= row base + kb*32 halves).
// Layout (matches B packing): lanes 0-15 hold K 0..15; lanes 16-31 hold K 16..31.
__device__ __forceinline__ v16h make_afrag16(const _Float16* rowp, int lane) {
  const int ch = (lane & 16) ? 16 : 0;
  v8h lo = *(const v8h*)(rowp + ch);
  v8h hi = *(const v8h*)(rowp + ch + 8);
  v16h a;
  #pragma unroll
  for (int e = 0; e < 8; ++e) { a[e] = lo[e]; a[e + 8] = hi[e]; }
  return a;
}

__device__ __forceinline__ v16h bload(const _Float16* sW, int frag, int lane) {
  return *(const v16h*)(sW + frag * 512 + lane * 16);
}

// Branch-free native-TRANS nonlinearities: v_exp_f32 (exp2) + v_rcp_f32.
#define LOG2E 1.4426950408889634f
__device__ __forceinline__ float fast_sig(float x) {
  return __builtin_amdgcn_rcpf(1.0f + __builtin_amdgcn_exp2f(-LOG2E * x));
}
__device__ __forceinline__ float fast_tanh(float x) {
  return 2.0f * __builtin_amdgcn_rcpf(1.0f + __builtin_amdgcn_exp2f(-2.0f * LOG2E * x)) - 1.0f;
}

// Gate GEMM: g[4dh x 16] (column-major, stride GSTR) = A1*W1 + A2*W2 + bias.
template<int NT, int KB1, int KB2>
__device__ __forceinline__ void layer_gates(const _Float16* rowA, int base1,
                                            const _Float16* rowB, int base2,
                                            const _Float16* __restrict__ sW,
                                            const float* __restrict__ sBias, int biasOff,
                                            float* __restrict__ gbuf, int lane) {
  const int col = lane & 15;
  const int rb  = (lane & 16) ? 8 : 0;
  v16h aA[KB1];
  #pragma unroll
  for (int kb = 0; kb < KB1; ++kb) aA[kb] = make_afrag16(rowA + kb * 32, lane);
  v16h aB[KB2 > 0 ? KB2 : 1];
  if (KB2 > 0) {
    #pragma unroll
    for (int kb = 0; kb < KB2; ++kb) aB[kb] = make_afrag16(rowB + kb * 32, lane);
  }
  #pragma unroll
  for (int nt = 0; nt < NT; ++nt) {
    const float b = sBias[biasOff + nt * 16 + col];
    v8f acc;
    #pragma unroll
    for (int r = 0; r < 8; ++r) acc[r] = b;
    #pragma unroll
    for (int kb = 0; kb < KB1; ++kb)
      acc = __builtin_amdgcn_wmma_f32_16x16x32_f16(false, aA[kb], false,
                bload(sW, base1 + kb * NT + nt, lane), (short)0, acc, false, false);
    #pragma unroll
    for (int kb = 0; kb < KB2; ++kb)
      acc = __builtin_amdgcn_wmma_f32_16x16x32_f16(false, aB[kb], false,
                bload(sW, base2 + kb * NT + nt, lane), (short)0, acc, false, false);
    // column-major store: C element (r,col) -> gbuf[(nt*16+col)*GSTR + rb + r]
    // = 8 consecutive dwords per lane -> two b128 stores, bank-conflict-free.
    float* gp = gbuf + (nt * 16 + col) * GSTR + rb;
    float4 s0 = { acc[0], acc[1], acc[2], acc[3] };
    float4 s1 = { acc[4], acc[5], acc[6], acc[7] };
    *(float4*)(gp)     = s0;
    *(float4*)(gp + 4) = s1;
  }
}

// Elementwise LSTM cell update (gate order i,f,g,o); cells enumerated as
// cid = u*16 + row  ->  row = cid & 15, u = cid >> 4 : no integer division,
// gate loads at constant offsets gate*DH*GSTR.
template<int DH, int HSTRIDE>
__device__ __forceinline__ void cell_update(const float* __restrict__ gbuf,
                                            float* __restrict__ cbuf,
                                            _Float16* __restrict__ hbuf, int lane) {
  for (int cid = lane; cid < 16 * DH; cid += 32) {
    const int row = cid & 15;
    const int u   = cid >> 4;
    const int g0  = u * GSTR + row;
    const float gi = gbuf[g0];
    const float gf = gbuf[g0 + DH * GSTR];
    const float gg = gbuf[g0 + 2 * DH * GSTR];
    const float go = gbuf[g0 + 3 * DH * GSTR];
    const float c  = cbuf[cid];
    const float cn = fast_sig(gf) * c + fast_sig(gi) * fast_tanh(gg);
    cbuf[cid]               = cn;
    hbuf[row * HSTRIDE + u] = (_Float16)(fast_sig(go) * fast_tanh(cn));
  }
}

// ---------------- main kernel ---------------------------------------------
// (128, 4): cap at <=256 VGPRs/wave so no s_set_vgpr_msb SALU traffic / spills.
__global__ __launch_bounds__(WAVES_PER_BLK * 32, 4)
void lstm_main(const float* __restrict__ x, const char* __restrict__ ws,
               float* __restrict__ out) {
  extern __shared__ char smem[];
  _Float16* sW    = (_Float16*)smem;
  float*    sBias = (float*)(smem + WS_BIAS_OFF);
  float*    sWl   = (float*)(smem + WS_WL_OFF);

  // stage packed weights (60 KB frags + biases + Wl) into LDS, shared by all waves
  {
    const uint4* src = (const uint4*)ws;
    uint4*       dst = (uint4*)smem;
    for (int i = threadIdx.x; i < WS_TOTAL / 16; i += blockDim.x) dst[i] = src[i];
  }
  __syncthreads();

  const int lane = threadIdx.x & 31;
  const int wv   = threadIdx.x >> 5;
  const int col  = lane & 15;

  // per-wave state region
  char*     pw  = smem + WS_TOTAL + wv * PW_BYTES;
  _Float16* xh1 = (_Float16*)pw;      // [16][32]: K0..6 = x_t, K7..21 = h1
  _Float16* h2  = xh1 + 512;          // [16][32]
  _Float16* h3  = h2 + 512;           // [16][32]
  _Float16* h4  = h3 + 512;           // [16][64]
  float*    c1  = (float*)(pw + 5120);        // 240 f (cid-indexed)
  float*    c2  = c1 + 240;                   // 320 f
  float*    c3  = c2 + 320;                   // 480 f
  float*    c4  = c3 + 480;                   // 640 f
  float*    gb  = c4 + 640;                   // [160][GSTR] gate scratch

  // zero f16 h-state (incl K-padding) + f32 c-state: (5120+6720)/4 u32
  {
    uint32_t* z = (uint32_t*)pw;
    for (int i = lane; i < 2960; i += 32) z[i] = 0u;
  }

  const int b0 = (blockIdx.x * WAVES_PER_BLK + wv) * 16;
  const float* xbase = x + (size_t)b0 * (T_STEPS * IN_DIM);
  float*       obase = out + (size_t)b0 * T_STEPS;

  const _Float16* rowXH1 = xh1 + col * 32;
  const _Float16* rowH2  = h2 + col * 32;
  const _Float16* rowH3  = h3 + col * 32;
  const _Float16* rowH4  = h4 + col * 64;

  for (int t = 0; t < T_STEPS; ++t) {
    // stage x_t (f32 global -> f16 LDS) into xh1[:, 0..6]; lane = row, no div
    if (lane < 16) {
      const float* xp = xbase + (size_t)lane * (T_STEPS * IN_DIM) + t * IN_DIM;
      _Float16*    xd = xh1 + lane * 32;
      #pragma unroll
      for (int k = 0; k < IN_DIM; ++k) xd[k] = (_Float16)xp[k];
    }

    // layer 1: single combined [x|h1] K-block, 4 WMMAs
    layer_gates<4, 1, 0>(rowXH1, FB_L1, nullptr, 0, sW, sBias, 0, gb, lane);
    cell_update<15, 32>(gb, c1, xh1 + 7, lane);        // h1 lives at xh1[:,7..21]

    // layer 2: same xh1 row (Wih2 packed K-shifted; x rows are zero)
    layer_gates<5, 1, 1>(rowXH1, FB_IH2, rowH2, FB_HH2, sW, sBias, 64, gb, lane);
    cell_update<20, 32>(gb, c2, h2, lane);

    // layer 3
    layer_gates<8, 1, 1>(rowH2, FB_IH3, rowH3, FB_HH3, sW, sBias, 144, gb, lane);
    cell_update<30, 32>(gb, c3, h3, lane);

    // layer 4 (hh K=40 -> 2 K-blocks)
    layer_gates<10, 1, 2>(rowH3, FB_IH4, rowH4, FB_HH4, sW, sBias, 272, gb, lane);
    cell_update<40, 64>(gb, c4, h4, lane);

    // final linear: out[b,t] = h4 . Wl + bl (one row per lane 0..15)
    if (lane < 16) {
      float acc = sWl[40];
      const _Float16* hr = h4 + lane * 64;
      #pragma unroll
      for (int k = 0; k < 40; ++k) acc = fmaf((float)hr[k], sWl[k], acc);
      obase[(size_t)lane * T_STEPS + t] = acc;
    }
  }
}

// ---------------- launch ---------------------------------------------------
extern "C" void kernel_launch(void* const* d_in, const int* in_sizes, int n_in,
                              void* d_out, int out_size, void* d_ws, size_t ws_size,
                              hipStream_t stream) {
  const float* x    = (const float*)d_in[0];
  const float* Wih1 = (const float*)d_in[1];
  const float* Whh1 = (const float*)d_in[2];
  const float* bih1 = (const float*)d_in[3];
  const float* bhh1 = (const float*)d_in[4];
  const float* Wih2 = (const float*)d_in[5];
  const float* Whh2 = (const float*)d_in[6];
  const float* bih2 = (const float*)d_in[7];
  const float* bhh2 = (const float*)d_in[8];
  const float* Wih3 = (const float*)d_in[9];
  const float* Whh3 = (const float*)d_in[10];
  const float* bih3 = (const float*)d_in[11];
  const float* bhh3 = (const float*)d_in[12];
  const float* Wih4 = (const float*)d_in[13];
  const float* Whh4 = (const float*)d_in[14];
  const float* bih4 = (const float*)d_in[15];
  const float* bhh4 = (const float*)d_in[16];
  const float* Wl   = (const float*)d_in[17];
  const float* bl   = (const float*)d_in[18];

  _Float16* frag    = (_Float16*)d_ws;
  float*    biasOut = (float*)((char*)d_ws + WS_BIAS_OFF);
  float*    wlOut   = (float*)((char*)d_ws + WS_WL_OFF);

  lstm_pack_frags<<<NFRAG, 32, 0, stream>>>(Wih1, Whh1, Wih2, Whh2,
                                            Wih3, Whh3, Wih4, Whh4, frag);
  lstm_pack_bias<<<1, 512, 0, stream>>>(bih1, bhh1, bih2, bhh2, bih3, bhh3,
                                        bih4, bhh4, Wl, bl, biasOut, wlOut);

  const int blocks = B_TOT / (WAVES_PER_BLK * 16);            // 128
  const size_t shmem = WS_TOTAL + WAVES_PER_BLK * PW_BYTES;   // ~158 KB
  lstm_main<<<blocks, WAVES_PER_BLK * 32, shmem, stream>>>(x, (const char*)d_ws,
                                                           (float*)d_out);
}
